// BasicBlock_45011257262579
// MI455X (gfx1250) — matrix operand order repaired and checked by reading, not compiled, not run on hardware
//
#include <hip/hip_runtime.h>
#include <hip/hip_bf16.h>

typedef __attribute__((ext_vector_type(16))) _Float16 v16h;
typedef __attribute__((ext_vector_type(8)))  _Float16 v8h;
typedef __attribute__((ext_vector_type(8)))  float    v8f;

#define BATCH  32
#define CH     128
#define HW     56
#define NSHIFT 9

// ---------------------------------------------------------------------------
// Prep: quantize weights to f16 integers, fold BN, precompute psum scales.
// wq[s][o][c] = round(clip(W[s][o][c]/a_w[s], -4, 3))   (exact small ints in f16)
// scale[s]   = a_w[s] / a_p    (psum quant:  a_p * clip(round(acc*scale), -4, 3))
// ---------------------------------------------------------------------------
__global__ void prep_kernel(const float* __restrict__ W1, const float* __restrict__ aw1,
                            const float* __restrict__ W2, const float* __restrict__ aw2,
                            const float* __restrict__ ap1, const float* __restrict__ ap2,
                            const float* __restrict__ g1, const float* __restrict__ b1,
                            const float* __restrict__ m1, const float* __restrict__ v1,
                            const float* __restrict__ g2, const float* __restrict__ b2,
                            const float* __restrict__ m2, const float* __restrict__ v2,
                            _Float16* __restrict__ qw1, _Float16* __restrict__ qw2,
                            float* __restrict__ scale1, float* __restrict__ scale2,
                            float* __restrict__ bninv1, float* __restrict__ bnbias1,
                            float* __restrict__ bninv2, float* __restrict__ bnbias2)
{
    const int i = blockIdx.x * blockDim.x + threadIdx.x;
    const int NW = NSHIFT * CH * CH;
    if (i < NW) {
        const int s = i / (CH * CH);
        float q1 = rintf(fminf(fmaxf(W1[i] / aw1[s], -4.f), 3.f));
        float q2 = rintf(fminf(fmaxf(W2[i] / aw2[s], -4.f), 3.f));
        qw1[i] = (_Float16)q1;
        qw2[i] = (_Float16)q2;
    }
    if (i < NSHIFT) {
        scale1[i] = aw1[i] / ap1[0];
        scale2[i] = aw2[i] / ap2[0];
    }
    if (i < CH) {
        float inv1 = g1[i] / sqrtf(v1[i] + 1e-5f);
        float inv2 = g2[i] / sqrtf(v2[i] + 1e-5f);
        bninv1[i]  = inv1;
        bnbias1[i] = b1[i] - m1[i] * inv1;
        bninv2[i]  = inv2;
        bnbias2[i] = b2[i] - m2[i] * inv2;
    }
}

// ---------------------------------------------------------------------------
// Fused split-conv3x3 (+ per-shift psum LSQ quant) + BN [+ residual] + ReLU.
// One block = (b, h, 16-wide w tile); 8 waves, each wave owns 16 out channels.
// Activations staged in LDS channel-contiguous f16 so each lane's WMMA
// B-fragment is one 32B contiguous ds load.
// ---------------------------------------------------------------------------
__global__ __launch_bounds__(256) void conv_block_kernel(
    const float* __restrict__ in, const _Float16* __restrict__ qw,
    const float* __restrict__ scale, const float* __restrict__ ap,
    const float* __restrict__ bninv, const float* __restrict__ bnbias,
    const float* __restrict__ residual, int use_res,
    float* __restrict__ out)
{
    // xs[(r*18 + col)*CH + c] : 3 halo rows x 18 halo cols x 128 channels (f16)
    __shared__ __align__(32) _Float16 xs[3 * 18 * CH];

    const int w0  = blockIdx.x * 16;   // w tile base (0,16,32,48)
    const int h0  = blockIdx.y;        // 0..55
    const int b   = blockIdx.z;        // 0..31
    const int tid = threadIdx.x;

    // ---- stage padded activation tile -> LDS (transposed, f16) ----
    for (int idx = tid; idx < 3 * CH * 18; idx += 256) {
        int r   = idx / (CH * 18);
        int rem = idx - r * (CH * 18);
        int c   = rem / 18;
        int col = rem - c * 18;
        int hh  = h0 - 1 + r;
        int ww  = w0 - 1 + col;
        float v = 0.f;
        if ((unsigned)hh < HW && (unsigned)ww < HW)
            v = in[(((size_t)b * CH + c) * HW + hh) * HW + ww];
        xs[(r * 18 + col) * CH + c] = (_Float16)v;
    }
    __syncthreads();

    const int wave   = tid >> 5;
    const int lane   = tid & 31;
    const int nloc   = lane & 15;          // N column of this lane
    const int hiHalf = lane >> 4;          // 0 or 1 (lane group)
    const int obase  = wave * 16;          // output-channel tile for this wave
    const int mrow   = obase + nloc;       // A-matrix row this lane loads
    const int akoff  = hiHalf * 16;        // A K sub-offset per lane group

    const float apv = ap[0];
    v8f outacc = {};

    for (int s = 0; s < NSHIFT; ++s) {
        const int dh = s % 3;              // SHIFTS[s] = (s%3, s/3)
        const int dw = s / 3;
        const _Float16* wrow = qw + ((size_t)(s * CH + mrow)) * CH;
        if (s + 1 < NSHIFT)
            __builtin_prefetch(qw + ((size_t)((s + 1) * CH + mrow)) * CH, 0, 1);

        v8f acc = {};
        #pragma unroll
        for (int kc = 0; kc < 4; ++kc) {
            const int k0 = kc * 32;
            // A fragment: lanes 0-15 hold K = k0+{0..15}; lanes 16-31: +16
            v16h a;
            *((v8h*)&a)       = *(const v8h*)(wrow + k0 + akoff);
            *(((v8h*)&a) + 1) = *(const v8h*)(wrow + k0 + 8 + akoff);
            // B fragment: lane group 0: K=k0..k0+15, group 1: K=k0+16..k0+31,
            // all at column nloc of the (dh,dw)-shifted view -> contiguous 32B
            const int kst = k0 + hiHalf * 16;
            v16h bf = *(const v16h*)(&xs[(dh * 18 + nloc + dw) * CH + kst]);
            acc = __builtin_amdgcn_wmma_f32_16x16x32_f16(
                      false, a, false, bf, (short)0, acc, false, false);
        }
        // per-shift LSQ psum quant: a_p * clip(round(a_w*acc / a_p), -4, 3)
        const float sc = scale[s];
        #pragma unroll
        for (int j = 0; j < 8; ++j) {
            float t = rintf(acc[j] * sc);
            t = fminf(fmaxf(t, -4.f), 3.f);
            outacc[j] += t * apv;
        }
    }

    // ---- epilogue: BN (+ residual) + ReLU + masked store ----
    // C/D layout: VGPR j, lanes 0-15 -> M=j ; lanes 16-31 -> M=j+8 ; N=lane&15
    const int moff = hiHalf * 8;
    const int w    = w0 + nloc;
    #pragma unroll
    for (int j = 0; j < 8; ++j) {
        const int o = obase + moff + j;
        if (w < HW) {
            const size_t gidx = (((size_t)b * CH + o) * HW + h0) * HW + w;
            float val = outacc[j] * bninv[o] + bnbias[o];
            if (use_res) val += residual[gidx];
            out[gidx] = fmaxf(val, 0.f);
        }
    }
}

// ---------------------------------------------------------------------------
extern "C" void kernel_launch(void* const* d_in, const int* in_sizes, int n_in,
                              void* d_out, int out_size, void* d_ws, size_t ws_size,
                              hipStream_t stream)
{
    const float* x   = (const float*)d_in[0];
    const float* W1  = (const float*)d_in[1];
    const float* aw1 = (const float*)d_in[2];
    const float* W2  = (const float*)d_in[3];
    const float* aw2 = (const float*)d_in[4];
    const float* ap1 = (const float*)d_in[5];
    const float* ap2 = (const float*)d_in[6];
    const float* g1  = (const float*)d_in[7];
    const float* b1  = (const float*)d_in[8];
    const float* m1  = (const float*)d_in[9];
    const float* v1  = (const float*)d_in[10];
    const float* g2  = (const float*)d_in[11];
    const float* b2  = (const float*)d_in[12];
    const float* m2  = (const float*)d_in[13];
    const float* v2  = (const float*)d_in[14];
    float* out = (float*)d_out;

    char* ws = (char*)d_ws;
    size_t off = 0;
    auto wsalloc = [&](size_t bytes) -> void* {
        void* p = ws + off;
        off += (bytes + 255) & ~(size_t)255;
        return p;
    };
    float*    y       = (float*)   wsalloc(sizeof(float) * (size_t)BATCH * CH * HW * HW);
    _Float16* qw1     = (_Float16*)wsalloc(sizeof(_Float16) * NSHIFT * CH * CH);
    _Float16* qw2     = (_Float16*)wsalloc(sizeof(_Float16) * NSHIFT * CH * CH);
    float*    scale1  = (float*)   wsalloc(sizeof(float) * 16);
    float*    scale2  = (float*)   wsalloc(sizeof(float) * 16);
    float*    bninv1  = (float*)   wsalloc(sizeof(float) * CH);
    float*    bnbias1 = (float*)   wsalloc(sizeof(float) * CH);
    float*    bninv2  = (float*)   wsalloc(sizeof(float) * CH);
    float*    bnbias2 = (float*)   wsalloc(sizeof(float) * CH);

    const int nprep = NSHIFT * CH * CH;
    prep_kernel<<<(nprep + 255) / 256, 256, 0, stream>>>(
        W1, aw1, W2, aw2, ap1, ap2,
        g1, b1, m1, v1, g2, b2, m2, v2,
        qw1, qw2, scale1, scale2, bninv1, bnbias1, bninv2, bnbias2);

    dim3 grid((HW + 15) / 16, HW, BATCH);   // (4, 56, 32)
    dim3 block(256);
    // layer 1: relu(bn1(split_conv(x, qw1)))
    conv_block_kernel<<<grid, block, 0, stream>>>(
        x, qw1, scale1, ap1, bninv1, bnbias1, nullptr, 0, y);
    // layer 2: relu(bn2(split_conv(y, qw2)) + x)
    conv_block_kernel<<<grid, block, 0, stream>>>(
        y, qw2, scale2, ap2, bninv2, bnbias2, x, 1, out);
}
